// Qwen3_5MoeSparseMoeBlock_79491254714412
// MI455X (gfx1250) — matrix operand order repaired and checked by reading, not compiled, not run on hardware
//
#include <hip/hip_runtime.h>
#include <math.h>

// ---------------- problem constants ----------------
constexpr int Td   = 1024;   // tokens (B*S)
constexpr int Hd   = 2048;   // hidden
constexpr int NE   = 32;     // experts
constexpr int Fd   = 768;    // expert ffn dim
constexpr int FSd  = 4096;   // shared ffn dim
constexpr int TOPK = 4;

// ---------------- WMMA types ----------------
typedef __attribute__((ext_vector_type(16))) __bf16 v16bf;
typedef __attribute__((ext_vector_type(8)))  float  v8f;

union AFrag { v16bf v; __bf16 h[16]; uint4 q[2]; };

__device__ __forceinline__ v8f wmma_bf16(v16bf a, v16bf b, v8f c) {
  // v_wmma_f32_16x16x32_bf16  (neg_a, A, neg_b, B, c_mod, C, reuse_a, reuse_b)
  return __builtin_amdgcn_wmma_f32_16x16x32_bf16(false, a, false, b, (short)0, c,
                                                 false, false);
}

__device__ __forceinline__ float fast_rcp(float x) {
  return __builtin_amdgcn_rcpf(x);          // v_rcp_f32
}
__device__ __forceinline__ float silu_mul(float g, float u) {
  return g * u * fast_rcp(1.0f + __expf(-g));
}
__device__ __forceinline__ float fast_sigmoid(float x) {
  return fast_rcp(1.0f + __expf(-x));
}

// A fragment (16x32, bf16): lane l holds row m=l&15, K chunks {c..c+7, c+16..c+23},
// c = (l>>4)*8.  Two 16-byte contiguous loads from a row-major bf16 row.
__device__ __forceinline__ v16bf load_a_tile(const __bf16* row, int k0, int lane) {
  int c = k0 + ((lane >> 4) << 3);
  AFrag a;
  a.q[0] = *(const uint4*)(row + c);
  a.q[1] = *(const uint4*)(row + c + 16);
  return a.v;
}

// B fragment from the pre-packed layout: each 32Kx16N tile is 512 bf16, stored so
// lane l's 16 values (n = n0+(l&15), K = k0+(l>=16?8:0)+{0..7,16..23}) occupy 32
// contiguous bytes at tileBase + l*16.  Two b128 loads, fully coalesced.
__device__ __forceinline__ v16bf load_b_packed(const __bf16* tileBase, int lane) {
  const __bf16* p = tileBase + lane * 16;
  AFrag b;
  b.q[0] = *(const uint4*)p;
  b.q[1] = *(const uint4*)(p + 8);
  return b.v;
}

// ---------------- weight pack: fp32 row-major (Kt x Nt) -> bf16 B-fragment tiles
// tile index = mat*(Kt/32)*(Nt/16) + nt*(Kt/32) + kt ; element = tile*512 + l*16 + j
__global__ __launch_bounds__(256) void k_pack(const float* __restrict__ W,
                                              __bf16* __restrict__ Wp,
                                              int Kt, int Nt, int total_tiles) {
  const int lane = threadIdx.x & 31;
  const int wave = threadIdx.x >> 5;
  int tile = blockIdx.x * 8 + wave;
  if (tile >= total_tiles) return;
  int ktiles = Kt >> 5;
  int tilesPerMat = ktiles * (Nt >> 4);
  int mat = tile / tilesPerMat;
  int r = tile - mat * tilesPerMat;
  int nt = r / ktiles;
  int kt = r - nt * ktiles;
  const float* src = W + (size_t)mat * Kt * Nt;
  int k0 = kt * 32 + ((lane >> 4) << 3);
  int n  = nt * 16 + (lane & 15);
  AFrag b;
#pragma unroll
  for (int j = 0; j < 16; ++j)
    b.h[j] = (__bf16)src[(size_t)(k0 + (j < 8 ? j : j + 8)) * Nt + n];
  __bf16* dst = Wp + (size_t)tile * 512 + lane * 16;
  *(uint4*)dst       = b.q[0];
  *(uint4*)(dst + 8) = b.q[1];
}

// ---------------- small utility kernels ----------------
__global__ void k_zero_counts(int* counts) {
  if (threadIdx.x < NE) counts[threadIdx.x] = 0;
}

__global__ void k_convert(const float* __restrict__ x, __bf16* __restrict__ xb, int n) {
  for (int i = blockIdx.x * blockDim.x + threadIdx.x; i < n;
       i += gridDim.x * blockDim.x)
    xb[i] = (__bf16)x[i];
}

// ---------------- router: logits -> softmax -> top4 -> renorm ----------------
__global__ __launch_bounds__(256) void k_router(const float* __restrict__ x,
                                                const float* __restrict__ rw,
                                                int* __restrict__ topi,
                                                float* __restrict__ topv,
                                                int* __restrict__ counts) {
  const int lane = threadIdx.x & 31;
  const int wave = threadIdx.x >> 5;
  const int t = blockIdx.x * 8 + wave;
  const float* xr = x + (size_t)t * Hd;
  const float* wr = rw + (size_t)lane * Hd;   // lane == expert (E==32==wave32)
  float acc = 0.f;
  for (int h = 0; h < Hd; ++h) acc += xr[h] * wr[h];

  // softmax across the 32 lanes
  float m = acc;
  for (int off = 16; off; off >>= 1) m = fmaxf(m, __shfl_xor(m, off, 32));
  float p = __expf(acc - m);
  float s = p;
  for (int off = 16; off; off >>= 1) s += __shfl_xor(s, off, 32);
  float prob = p * fast_rcp(s);

  // iterative top-4 (value, index) butterfly reduction
  float kv[TOPK]; int ki[TOPK];
  float cur = prob;
#pragma unroll
  for (int it = 0; it < TOPK; ++it) {
    float v = cur; int idx = lane;
    for (int off = 16; off; off >>= 1) {
      float ov = __shfl_xor(v, off, 32);
      int   oi = __shfl_xor(idx, off, 32);
      if (ov > v || (ov == v && oi < idx)) { v = ov; idx = oi; }
    }
    kv[it] = v; ki[it] = idx;
    if (lane == idx) cur = -1.0f;
  }
  if (lane == 0) {
    float rs = fast_rcp(kv[0] + kv[1] + kv[2] + kv[3]);
#pragma unroll
    for (int k = 0; k < TOPK; ++k) {
      topi[t * TOPK + k] = ki[k];
      topv[t * TOPK + k] = kv[k] * rs;
      atomicAdd(&counts[ki[k]], 1);
    }
  }
}

__global__ void k_scan(const int* __restrict__ counts, int* __restrict__ offs,
                       int* __restrict__ cursors) {
  if (threadIdx.x == 0) {
    int o = 0;
    for (int e = 0; e < NE; ++e) { offs[e] = o; o += counts[e]; cursors[e] = 0; }
  }
}

__global__ void k_fill(const int* __restrict__ topi, const float* __restrict__ topv,
                       const int* __restrict__ offs, int* __restrict__ cursors,
                       int* __restrict__ token_list, float* __restrict__ weight_list) {
  int t = blockIdx.x * blockDim.x + threadIdx.x;
  if (t >= Td) return;
#pragma unroll
  for (int k = 0; k < TOPK; ++k) {
    int e = topi[t * TOPK + k];
    int pos = atomicAdd(&cursors[e], 1);
    int slot = offs[e] + pos;
    token_list[slot] = t;
    weight_list[slot] = topv[t * TOPK + k];
  }
}

// ---------------- expert gate/up:  act = silu(x Wg) * (x Wu)  ----------------
// 2 M-tiles (32 rows) per iteration: B fragments loaded once, 4 independent
// WMMA accumulator chains for ILP.  B comes from packed bf16 tiles (2 b128/frag).
__global__ __launch_bounds__(256) void k_expert_gateup(
    const __bf16* __restrict__ xb, const __bf16* __restrict__ pwg,
    const __bf16* __restrict__ pwu, const int* __restrict__ offs,
    const int* __restrict__ counts, const int* __restrict__ token_list,
    __bf16* __restrict__ act) {
  const int lane = threadIdx.x & 31;
  const int wave = threadIdx.x >> 5;
  const int e  = blockIdx.x / (Fd / 128);
  const int fb = blockIdx.x % (Fd / 128);
  const int n0 = fb * 128 + wave * 16;
  const int seg = offs[e], cnt = counts[e];
  const __bf16* wgp = pwg + (size_t)e * Hd * Fd;
  const __bf16* wup = pwu + (size_t)e * Hd * Fd;
  const size_t tileRow = (size_t)(n0 >> 4) * (Hd >> 5);   // nt * ktiles
  const int n = n0 + (lane & 15);
  const int half = lane >> 4;

  for (int m0 = 0; m0 < cnt; m0 += 32) {
    int r0 = m0 + (lane & 15);
    int r1 = r0 + 16;
    int tok0 = token_list[seg + min(r0, cnt - 1)];  // clamp: invalid rows discarded
    int tok1 = token_list[seg + min(r1, cnt - 1)];
    const __bf16* arow0 = xb + (size_t)tok0 * Hd;
    const __bf16* arow1 = xb + (size_t)tok1 * Hd;
    v8f cg0 = {}, cu0 = {}, cg1 = {}, cu1 = {};
    for (int k0 = 0; k0 < Hd; k0 += 32) {
      size_t toff = (tileRow + (k0 >> 5)) * 512;
      v16bf a0 = load_a_tile(arow0, k0, lane);
      v16bf a1 = load_a_tile(arow1, k0, lane);
      v16bf bg = load_b_packed(wgp + toff, lane);
      v16bf bu = load_b_packed(wup + toff, lane);
      cg0 = wmma_bf16(a0, bg, cg0);
      cg1 = wmma_bf16(a1, bg, cg1);
      cu0 = wmma_bf16(a0, bu, cu0);
      cu1 = wmma_bf16(a1, bu, cu1);
    }
#pragma unroll
    for (int i = 0; i < 8; ++i) {
      int rr = m0 + i + half * 8;
      if (rr < cnt)
        act[(size_t)(seg + rr) * Fd + n] = (__bf16)silu_mul(cg0[i], cu0[i]);
      int rr1 = rr + 16;
      if (rr1 < cnt)
        act[(size_t)(seg + rr1) * Fd + n] = (__bf16)silu_mul(cg1[i], cu1[i]);
    }
  }
}

// ---------------- expert down:  out[t] += w * (act Wd) -----------------------
__global__ __launch_bounds__(256) void k_expert_down(
    const __bf16* __restrict__ act, const __bf16* __restrict__ pwd,
    const int* __restrict__ offs, const int* __restrict__ counts,
    const int* __restrict__ token_list, const float* __restrict__ weight_list,
    float* __restrict__ out) {
  const int lane = threadIdx.x & 31;
  const int wave = threadIdx.x >> 5;
  const int e  = blockIdx.x / (Hd / 128);
  const int hb = blockIdx.x % (Hd / 128);
  const int n0 = hb * 128 + wave * 16;
  const int seg = offs[e], cnt = counts[e];
  const __bf16* wdp = pwd + (size_t)e * Fd * Hd;
  const size_t tileRow = (size_t)(n0 >> 4) * (Fd >> 5);
  const int n = n0 + (lane & 15);
  const int half = lane >> 4;

  for (int m0 = 0; m0 < cnt; m0 += 32) {
    int r0 = m0 + (lane & 15);
    int r1 = r0 + 16;
    const __bf16* arow0 = act + (size_t)(seg + min(r0, cnt - 1)) * Fd;
    const __bf16* arow1 = act + (size_t)(seg + min(r1, cnt - 1)) * Fd;
    v8f c0 = {}, c1 = {};
    for (int k0 = 0; k0 < Fd; k0 += 32) {
      v16bf a0 = load_a_tile(arow0, k0, lane);
      v16bf a1 = load_a_tile(arow1, k0, lane);
      v16bf b  = load_b_packed(wdp + (tileRow + (k0 >> 5)) * 512, lane);
      c0 = wmma_bf16(a0, b, c0);
      c1 = wmma_bf16(a1, b, c1);
    }
#pragma unroll
    for (int i = 0; i < 8; ++i) {
      int rr = m0 + i + half * 8;
      if (rr < cnt) {
        int slot = seg + rr;
        atomicAdd(&out[(size_t)token_list[slot] * Hd + n],
                  weight_list[slot] * c0[i]);
      }
      int rr1 = rr + 16;
      if (rr1 < cnt) {
        int slot = seg + rr1;
        atomicAdd(&out[(size_t)token_list[slot] * Hd + n],
                  weight_list[slot] * c1[i]);
      }
    }
  }
}

// ---------------- shared expert ----------------------------------------------
__global__ __launch_bounds__(256) void k_sgate(const float* __restrict__ x,
                                               const float* __restrict__ sg,
                                               float* __restrict__ sgate) {
  const int lane = threadIdx.x & 31;
  const int wave = threadIdx.x >> 5;
  const int t = blockIdx.x * 8 + wave;
  float acc = 0.f;
  for (int h = lane; h < Hd; h += 32) acc += x[(size_t)t * Hd + h] * sg[h];
  for (int off = 16; off; off >>= 1) acc += __shfl_xor(acc, off, 32);
  if (lane == 0) sgate[t] = fast_sigmoid(acc);
}

__global__ __launch_bounds__(256) void k_shared_gateup(
    const __bf16* __restrict__ xb, const __bf16* __restrict__ pswg,
    const __bf16* __restrict__ pswu, __bf16* __restrict__ y1) {
  const int lane = threadIdx.x & 31;
  const int wave = threadIdx.x >> 5;
  const int n0 = blockIdx.x * 128 + wave * 16;
  const size_t tileRow = (size_t)(n0 >> 4) * (Hd >> 5);
  const int n = n0 + (lane & 15);
  const int half = lane >> 4;
  for (int m0 = 0; m0 < Td; m0 += 32) {
    const __bf16* arow0 = xb + (size_t)(m0 + (lane & 15)) * Hd;
    const __bf16* arow1 = arow0 + (size_t)16 * Hd;
    v8f cg0 = {}, cu0 = {}, cg1 = {}, cu1 = {};
    for (int k0 = 0; k0 < Hd; k0 += 32) {
      size_t toff = (tileRow + (k0 >> 5)) * 512;
      v16bf a0 = load_a_tile(arow0, k0, lane);
      v16bf a1 = load_a_tile(arow1, k0, lane);
      v16bf bg = load_b_packed(pswg + toff, lane);
      v16bf bu = load_b_packed(pswu + toff, lane);
      cg0 = wmma_bf16(a0, bg, cg0);
      cg1 = wmma_bf16(a1, bg, cg1);
      cu0 = wmma_bf16(a0, bu, cu0);
      cu1 = wmma_bf16(a1, bu, cu1);
    }
#pragma unroll
    for (int i = 0; i < 8; ++i) {
      int t0 = m0 + i + half * 8;
      y1[(size_t)t0 * FSd + n]        = (__bf16)silu_mul(cg0[i], cu0[i]);
      y1[(size_t)(t0 + 16) * FSd + n] = (__bf16)silu_mul(cg1[i], cu1[i]);
    }
  }
}

// writes (initializes) out = sigmoid_gate * (y1 @ sw_down)
__global__ __launch_bounds__(256) void k_shared_down(
    const __bf16* __restrict__ y1, const __bf16* __restrict__ pswd,
    const float* __restrict__ sgate, float* __restrict__ out) {
  const int lane = threadIdx.x & 31;
  const int wave = threadIdx.x >> 5;
  const int n0 = blockIdx.x * 128 + wave * 16;
  const size_t tileRow = (size_t)(n0 >> 4) * (FSd >> 5);
  const int n = n0 + (lane & 15);
  const int half = lane >> 4;
  for (int m0 = 0; m0 < Td; m0 += 32) {
    const __bf16* arow0 = y1 + (size_t)(m0 + (lane & 15)) * FSd;
    const __bf16* arow1 = arow0 + (size_t)16 * FSd;
    v8f c0 = {}, c1 = {};
    for (int k0 = 0; k0 < FSd; k0 += 32) {
      v16bf a0 = load_a_tile(arow0, k0, lane);
      v16bf a1 = load_a_tile(arow1, k0, lane);
      v16bf b  = load_b_packed(pswd + (tileRow + (k0 >> 5)) * 512, lane);
      c0 = wmma_bf16(a0, b, c0);
      c1 = wmma_bf16(a1, b, c1);
    }
#pragma unroll
    for (int i = 0; i < 8; ++i) {
      int t0 = m0 + i + half * 8;
      out[(size_t)t0 * Hd + n]        = sgate[t0]      * c0[i];
      out[(size_t)(t0 + 16) * Hd + n] = sgate[t0 + 16] * c1[i];
    }
  }
}

// ---------------- workspace layout (bytes) ----------------
constexpr size_t OFF_XB     = 0;                                     // bf16 x  [T,H]
constexpr size_t OFF_ACT    = OFF_XB   + (size_t)Td * Hd * 2;        // bf16 act[T*K,F]
constexpr size_t OFF_Y1     = OFF_ACT  + (size_t)Td * TOPK * Fd * 2; // bf16 y1 [T,FS]
constexpr size_t OFF_SGATE  = OFF_Y1   + (size_t)Td * FSd * 2;       // f32  [T]
constexpr size_t OFF_TOPI   = OFF_SGATE + (size_t)Td * 4;            // i32  [T,K]
constexpr size_t OFF_TOPV   = OFF_TOPI + (size_t)Td * TOPK * 4;      // f32  [T,K]
constexpr size_t OFF_TLIST  = OFF_TOPV + (size_t)Td * TOPK * 4;      // i32  [T*K]
constexpr size_t OFF_WLIST  = OFF_TLIST + (size_t)Td * TOPK * 4;     // f32  [T*K]
constexpr size_t OFF_CNT    = OFF_WLIST + (size_t)Td * TOPK * 4;     // i32  [E]
constexpr size_t OFF_OFFS   = OFF_CNT  + 256;                        // i32  [E]
constexpr size_t OFF_CURS   = OFF_OFFS + 256;                        // i32  [E]
// packed bf16 weights (B-fragment tile layout)
constexpr size_t OFF_PWG  = OFF_CURS + 256;                          // [E,H,F]
constexpr size_t OFF_PWU  = OFF_PWG  + (size_t)NE * Hd * Fd * 2;
constexpr size_t OFF_PWD  = OFF_PWU  + (size_t)NE * Hd * Fd * 2;     // [E,F,H]
constexpr size_t OFF_PSWG = OFF_PWD  + (size_t)NE * Fd * Hd * 2;     // [H,FS]
constexpr size_t OFF_PSWU = OFF_PSWG + (size_t)Hd * FSd * 2;
constexpr size_t OFF_PSWD = OFF_PSWU + (size_t)Hd * FSd * 2;         // [FS,H]

extern "C" void kernel_launch(void* const* d_in, const int* in_sizes, int n_in,
                              void* d_out, int out_size, void* d_ws, size_t ws_size,
                              hipStream_t stream) {
  const float* x    = (const float*)d_in[0];  // [1,1024,2048]
  const float* rw   = (const float*)d_in[1];  // [32,2048]
  const float* wg   = (const float*)d_in[2];  // [32,2048,768]
  const float* wu   = (const float*)d_in[3];  // [32,2048,768]
  const float* wd   = (const float*)d_in[4];  // [32,768,2048]
  const float* swg  = (const float*)d_in[5];  // [2048,4096]
  const float* swu  = (const float*)d_in[6];  // [2048,4096]
  const float* swd  = (const float*)d_in[7];  // [4096,2048]
  const float* sgw  = (const float*)d_in[8];  // [2048,1]
  float* out = (float*)d_out;                 // [1,1024,2048]

  uint8_t* ws = (uint8_t*)d_ws;
  __bf16* xb     = (__bf16*)(ws + OFF_XB);
  __bf16* act    = (__bf16*)(ws + OFF_ACT);
  __bf16* y1     = (__bf16*)(ws + OFF_Y1);
  float*  sgate  = (float*) (ws + OFF_SGATE);
  int*    topi   = (int*)   (ws + OFF_TOPI);
  float*  topv   = (float*) (ws + OFF_TOPV);
  int*    tlist  = (int*)   (ws + OFF_TLIST);
  float*  wlist  = (float*) (ws + OFF_WLIST);
  int*    counts = (int*)   (ws + OFF_CNT);
  int*    offs   = (int*)   (ws + OFF_OFFS);
  int*    curs   = (int*)   (ws + OFF_CURS);
  __bf16* pwg    = (__bf16*)(ws + OFF_PWG);
  __bf16* pwu    = (__bf16*)(ws + OFF_PWU);
  __bf16* pwd    = (__bf16*)(ws + OFF_PWD);
  __bf16* pswg   = (__bf16*)(ws + OFF_PSWG);
  __bf16* pswu   = (__bf16*)(ws + OFF_PSWU);
  __bf16* pswd   = (__bf16*)(ws + OFF_PSWD);

  // ---- weight packing (fp32 -> bf16 WMMA-B tiles), one-time stream ops ----
  {
    int tg = NE * (Hd / 32) * (Fd / 16);     // w_gate / w_up tiles
    int td = NE * (Fd / 32) * (Hd / 16);     // w_down tiles
    int ts = (Hd / 32) * (FSd / 16);         // sw_gate / sw_up tiles
    int td2 = (FSd / 32) * (Hd / 16);        // sw_down tiles
    k_pack<<<(tg + 7) / 8, 256, 0, stream>>>(wg, pwg, Hd, Fd, tg);
    k_pack<<<(tg + 7) / 8, 256, 0, stream>>>(wu, pwu, Hd, Fd, tg);
    k_pack<<<(td + 7) / 8, 256, 0, stream>>>(wd, pwd, Fd, Hd, td);
    k_pack<<<(ts + 7) / 8, 256, 0, stream>>>(swg, pswg, Hd, FSd, ts);
    k_pack<<<(ts + 7) / 8, 256, 0, stream>>>(swu, pswu, Hd, FSd, ts);
    k_pack<<<(td2 + 7) / 8, 256, 0, stream>>>(swd, pswd, FSd, Hd, td2);
  }

  // ---- routing pipeline ----
  k_zero_counts<<<1, 64, 0, stream>>>(counts);
  k_convert<<<512, 256, 0, stream>>>(x, xb, Td * Hd);
  k_router<<<Td / 8, 256, 0, stream>>>(x, rw, topi, topv, counts);
  k_scan<<<1, 32, 0, stream>>>(counts, offs, curs);
  k_fill<<<Td / 256, 256, 0, stream>>>(topi, topv, offs, curs, tlist, wlist);

  // ---- shared expert (k_shared_down initializes d_out with '=') ----
  k_sgate<<<Td / 8, 256, 0, stream>>>(x, sgw, sgate);
  k_shared_gateup<<<FSd / 128, 256, 0, stream>>>(xb, pswg, pswu, y1);
  k_shared_down<<<Hd / 128, 256, 0, stream>>>(y1, pswd, sgate, out);

  // ---- routed experts (k_expert_down accumulates with atomicAdd) ----
  k_expert_gateup<<<NE * (Fd / 128), 256, 0, stream>>>(xb, pwg, pwu, offs, counts,
                                                       tlist, act);
  k_expert_down<<<NE * (Hd / 128), 256, 0, stream>>>(act, pwd, offs, counts, tlist,
                                                     wlist, out);
  (void)in_sizes; (void)n_in; (void)out_size; (void)ws_size;
}
